// VisionMambaBlock_55336358642833
// MI455X (gfx1250) — matrix operand order repaired and checked by reading, not compiled, not run on hardware
//
#include <hip/hip_runtime.h>

typedef __attribute__((ext_vector_type(16))) _Float16 v16h;
typedef __attribute__((ext_vector_type(8)))  float    v8f;

#define BSZ     4
#define SEQ     1024
#define DM      768
#define BL      (BSZ*SEQ)            // 4096 rows
#define DSTATE  16
#define DTRANK  48
#define XDBL_N  (DTRANK + 2*DSTATE)  // 80
#define NHEADS  8
#define DK      96
#define MLPH    3072

#if __has_builtin(__builtin_amdgcn_global_load_async_to_lds_b128)
#define USE_ASYNC_LDS 1
#else
#define USE_ASYNC_LDS 0
#endif
#define AS1 __attribute__((address_space(1)))
#define AS3 __attribute__((address_space(3)))
typedef int v4i __attribute__((vector_size(16)));   // native vector, per diagnostic
typedef AS1 v4i gint4_t;   // global-space int4 (builtin param 0)
typedef AS3 v4i lint4_t;   // LDS-space int4    (builtin param 1)

// ---------------------------------------------------------------------------
// GEMM, f16 operands in global memory:
//   C(MxN) = epi( alpha * A(MxK) @ B^T + bias ) [+ resid]
// A: row-major MxK (lda). B: row-major NxK (ldb).
// Caller guarantees: M % 128 == 0, K % 32 == 0, lda/ldb % 8 == 0; B rows may
// be over-read up to 128 (store guard drops those columns); N % 16 == 0.
// EPI: 0=none 1=+bias 2=+bias,softplus 3=+bias,gelu 4=+bias,+resid
// OUT: 0=f32  1=f16  2=both  3=f16 transposed-per-head (for V)
// ---------------------------------------------------------------------------
constexpr int BM = 128, BN = 128, BK = 32, SAK = 40;   // SAK: padded K stride
constexpr int TSZ = BM * SAK;                          // halves per tile buffer

union FragH { v16h v; unsigned int u[8]; _Float16 h[16]; };

template <int EPI, int OUT>
__global__ __launch_bounds__(256) void gemm_wmma_k(
    const _Float16* __restrict__ A, const _Float16* __restrict__ B,
    float* __restrict__ Cf, _Float16* __restrict__ Ch,
    const float* __restrict__ bias, const float* __restrict__ resid,
    int N, int K, int lda, int ldb, int ldc, float alpha) {
  const int tid  = threadIdx.x;
  const int lane = tid & 31;
  const int wv   = tid >> 5;
  const int wm   = wv >> 1;          // 0..3 (32-row strip)
  const int wn   = wv & 1;           // 0..1 (64-col strip)
  const int hi   = lane >> 4;
  const int lq   = lane & 15;
  const int m0   = blockIdx.y * BM;
  const int n0   = blockIdx.x * BN;

  // staging: 512 chunks of 8 halves per tile, 2 per thread
  const int c0 = tid, c1 = tid + 256;
  const int r0c = c0 >> 2, k0c = (c0 & 3) * 8;
  const int r1c = c1 >> 2, k1c = (c1 & 3) * 8;

  const _Float16* Ar0 = A + (long)(m0 + r0c) * lda + k0c;
  const _Float16* Ar1 = A + (long)(m0 + r1c) * lda + k1c;
  const _Float16* Br0 = B + (long)(n0 + r0c) * ldb + k0c;
  const _Float16* Br1 = B + (long)(n0 + r1c) * ldb + k1c;

  v8f acc[2][4] = {};
  const int S = K / BK;

#if USE_ASYNC_LDS
  // ---- async copy engine staging, triple-buffered LDS, 1 barrier/stage ----
  __shared__ _Float16 As[3 * TSZ];
  __shared__ _Float16 Bs[3 * TSZ];
  auto issue = [&](int st) {
    int kb = st * BK;
    _Float16* as = As + (st % 3) * TSZ;
    _Float16* bs = Bs + (st % 3) * TSZ;
    __builtin_amdgcn_global_load_async_to_lds_b128(
        (gint4_t*)(Ar0 + kb), (lint4_t*)&as[r0c * SAK + k0c], 0, 0);
    __builtin_amdgcn_global_load_async_to_lds_b128(
        (gint4_t*)(Ar1 + kb), (lint4_t*)&as[r1c * SAK + k1c], 0, 0);
    __builtin_amdgcn_global_load_async_to_lds_b128(
        (gint4_t*)(Br0 + kb), (lint4_t*)&bs[r0c * SAK + k0c], 0, 0);
    __builtin_amdgcn_global_load_async_to_lds_b128(
        (gint4_t*)(Br1 + kb), (lint4_t*)&bs[r1c * SAK + k1c], 0, 0);
  };
  issue(0);
  for (int s = 0; s < S; ++s) {
    if (s + 1 < S) {
      issue(s + 1);                                     // into buf (s+1)%3
      asm volatile("s_wait_asynccnt 0x4" ::: "memory"); // stage s retired
    } else {
      asm volatile("s_wait_asynccnt 0x0" ::: "memory");
    }
    __syncthreads();                                    // cross-wave visibility
    const _Float16* as = As + (s % 3) * TSZ;
    const _Float16* bs = Bs + (s % 3) * TSZ;
#else
  // ---- register staging, double-buffered LDS ----
  __shared__ _Float16 As[2 * TSZ];
  __shared__ _Float16 Bs[2 * TSZ];
  int4 ra0 = *(const int4*)Ar0;
  int4 ra1 = *(const int4*)Ar1;
  int4 rb0 = *(const int4*)Br0;
  int4 rb1 = *(const int4*)Br1;
  for (int s = 0; s < S; ++s) {
    _Float16* as = As + (s & 1) * TSZ;
    _Float16* bs = Bs + (s & 1) * TSZ;
    *(int4*)&as[r0c * SAK + k0c] = ra0;
    *(int4*)&as[r1c * SAK + k1c] = ra1;
    *(int4*)&bs[r0c * SAK + k0c] = rb0;
    *(int4*)&bs[r1c * SAK + k1c] = rb1;
    __syncthreads();
    if (s + 1 < S) {                  // issue next-stage global loads early
      int kb = (s + 1) * BK;
      ra0 = *(const int4*)(Ar0 + kb);
      ra1 = *(const int4*)(Ar1 + kb);
      rb0 = *(const int4*)(Br0 + kb);
      rb1 = *(const int4*)(Br1 + kb);
    }
#endif

    FragH af[2], bf[4];
#pragma unroll
    for (int ti = 0; ti < 2; ++ti) {
      int row = wm * 32 + ti * 16 + lq;
#pragma unroll
      for (int j = 0; j < 8; ++j) {
        int kb = 16 * hi + 2 * j;   // A frag K(lane,pair)
        af[ti].u[j] = *(const unsigned int*)&as[row * SAK + kb];
      }
    }
#pragma unroll
    for (int tj = 0; tj < 4; ++tj) {
      int col = wn * 64 + tj * 16 + lq;
#pragma unroll
      for (int j = 0; j < 8; ++j)                        // B frag K = 16*hi+2j
        bf[tj].u[j] = *(const unsigned int*)&bs[col * SAK + 16 * hi + 2 * j];
    }
#pragma unroll
    for (int ti = 0; ti < 2; ++ti)
#pragma unroll
      for (int tj = 0; tj < 4; ++tj)
        acc[ti][tj] = __builtin_amdgcn_wmma_f32_16x16x32_f16(
            false, af[ti].v, false, bf[tj].v, (short)0, acc[ti][tj], false, false);
  }

  // epilogue (C/D layout: vgpr i -> M = i + 8*hi, N = lane%16)
#pragma unroll
  for (int ti = 0; ti < 2; ++ti)
#pragma unroll
    for (int tj = 0; tj < 4; ++tj) {
      int ntile = n0 + wn * 64 + tj * 16;      // uniform per-tile store guard
      if (ntile >= N) continue;
      int colg = ntile + lq;
#pragma unroll
      for (int i = 0; i < 8; ++i) {
        int rowg = m0 + wm * 32 + ti * 16 + 8 * hi + i;
        float x = acc[ti][tj][i] * alpha;
        if (EPI >= 1) x += bias[colg];
        if (EPI == 2) x = (x > 20.0f) ? x : log1pf(expf(x));
        if (EPI == 3) x = 0.5f * x * (1.0f + erff(x * 0.7071067811865476f));
        if (EPI == 4) x += resid[(long)rowg * ldc + colg];
        if (OUT == 0 || OUT == 2) Cf[(long)rowg * ldc + colg] = x;
        if (OUT == 1 || OUT == 2) Ch[(long)rowg * ldc + colg] = (_Float16)x;
        if (OUT == 3) {            // per-head transposed V: vT[b][h][j][m]
          int b = rowg >> 10, m = rowg & (SEQ - 1);
          int h = colg / DK, j = colg % DK;
          Ch[(((long)b * NHEADS + h) * DK + j) * SEQ + m] = (_Float16)x;
        }
      }
    }
}

// ---------------------------------------------------------------------------
// weight convert + transpose with K zero-padding:
//   o[n*Kp + k] = (k < K) ? (f16) w[k*N + n] : 0      (o is N x Kp)
// ---------------------------------------------------------------------------
__global__ void cvtT_k(const float* __restrict__ w, _Float16* __restrict__ o,
                       int K, int N, int Kp) {
  int i = blockIdx.x * 256 + threadIdx.x;
  if (i >= N * Kp) return;
  int n = i / Kp, k = i % Kp;
  o[i] = (k < K) ? (_Float16)w[(long)k * N + n] : (_Float16)0.0f;
}

// ---------------------------------------------------------------------------
// LayerNorm over DM=768, one block per row; writes f32 + f16 copies
// ---------------------------------------------------------------------------
__global__ __launch_bounds__(256) void ln_k(const float* __restrict__ x,
                                            const float* __restrict__ w,
                                            const float* __restrict__ b,
                                            float* __restrict__ out,
                                            _Float16* __restrict__ out16) {
  const int row = blockIdx.x;
  const float* p = x + (long)row * DM;
  float v[3], s = 0.f, sq = 0.f;
#pragma unroll
  for (int j = 0; j < 3; ++j) {
    float t = p[threadIdx.x + j * 256];
    v[j] = t; s += t; sq += t * t;
  }
  __shared__ float r1[256], r2[256];
  r1[threadIdx.x] = s; r2[threadIdx.x] = sq;
  __syncthreads();
  for (int off = 128; off; off >>= 1) {
    if (threadIdx.x < off) {
      r1[threadIdx.x] += r1[threadIdx.x + off];
      r2[threadIdx.x] += r2[threadIdx.x + off];
    }
    __syncthreads();
  }
  float mean = r1[0] * (1.0f / DM);
  float var  = r2[0] * (1.0f / DM) - mean * mean;
  float rstd = rsqrtf(var + 1e-5f);
#pragma unroll
  for (int j = 0; j < 3; ++j) {
    int i = threadIdx.x + j * 256;
    float y = (v[j] - mean) * rstd * w[i] + b[i];
    out[(long)row * DM + i] = y;
    out16[(long)row * DM + i] = (_Float16)y;
  }
}

// ---------------------------------------------------------------------------
// causal depthwise conv1d K=4; writes f32 + f16
// ---------------------------------------------------------------------------
__global__ void conv_k(const float* __restrict__ xn, const float* __restrict__ w,
                       const float* __restrict__ b, float* __restrict__ out,
                       _Float16* __restrict__ out16) {
  int i = blockIdx.x * 256 + threadIdx.x;
  if (i >= BL * DM) return;
  int c  = i % DM;
  int l  = (i / DM) % SEQ;
  int bb = i / (DM * SEQ);
  float acc = b[c];
#pragma unroll
  for (int t = 0; t < 4; ++t) {
    int ll = l - 3 + t;
    if (ll >= 0) acc += w[c * 4 + t] * xn[((long)bb * SEQ + ll) * DM + c];
  }
  out[i] = acc;
  out16[i] = (_Float16)acc;
}

// ---------------------------------------------------------------------------
// selective scan: thread = (batch, channel); 16 states in registers
// ---------------------------------------------------------------------------
__global__ __launch_bounds__(256) void scan_k(
    const float* __restrict__ A_log, const float* __restrict__ Dp,
    const float* __restrict__ xdbl,  const float* __restrict__ dt,
    const float* __restrict__ xconv, const float* __restrict__ x,
    float* __restrict__ out) {
  const int c = blockIdx.x * 256 + threadIdx.x;
  const int b = blockIdx.y;
  __shared__ float BmC[32];
  float Ar[DSTATE], h[DSTATE];
#pragma unroll
  for (int n = 0; n < DSTATE; ++n) { Ar[n] = -expf(A_log[c * DSTATE + n]); h[n] = 0.f; }
  const float dpc = Dp[c];
  for (int l = 0; l < SEQ; ++l) {
    if (threadIdx.x < 32)
      BmC[threadIdx.x] = xdbl[((long)b * SEQ + l) * XDBL_N + DTRANK + threadIdx.x];
    __syncthreads();
    long idx = ((long)b * SEQ + l) * DM + c;
    float dtv = dt[idx], xc = xconv[idx];
    float y = 0.f;
#pragma unroll
    for (int n = 0; n < DSTATE; ++n) {
      h[n] = expf(dtv * Ar[n]) * h[n] + dtv * BmC[n] * xc;
      y += h[n] * BmC[16 + n];
    }
    out[idx] = x[idx] + y + xc * dpc;
    __syncthreads();
  }
}

// ---------------------------------------------------------------------------
// row softmax over SEQ, writes f16 attention matrix
// ---------------------------------------------------------------------------
__global__ __launch_bounds__(256) void softmax_k(const float* __restrict__ s,
                                                 _Float16* __restrict__ o) {
  const float* p = s + (long)blockIdx.x * SEQ;
  _Float16*   q = o + (long)blockIdx.x * SEQ;
  __shared__ float red[256];
  float v[4], mx = -3.0e38f;
#pragma unroll
  for (int j = 0; j < 4; ++j) { v[j] = p[threadIdx.x + j * 256]; mx = fmaxf(mx, v[j]); }
  red[threadIdx.x] = mx;
  __syncthreads();
  for (int off = 128; off; off >>= 1) {
    if (threadIdx.x < off) red[threadIdx.x] = fmaxf(red[threadIdx.x], red[threadIdx.x + off]);
    __syncthreads();
  }
  mx = red[0];
  __syncthreads();
  float sum = 0.f;
#pragma unroll
  for (int j = 0; j < 4; ++j) { v[j] = expf(v[j] - mx); sum += v[j]; }
  red[threadIdx.x] = sum;
  __syncthreads();
  for (int off = 128; off; off >>= 1) {
    if (threadIdx.x < off) red[threadIdx.x] += red[threadIdx.x + off];
    __syncthreads();
  }
  float inv = 1.0f / red[0];
#pragma unroll
  for (int j = 0; j < 4; ++j) q[threadIdx.x + j * 256] = (_Float16)(v[j] * inv);
}

// ---------------------------------------------------------------------------
__global__ void combine_k(const float* __restrict__ ssm, const float* __restrict__ att,
                          const float* __restrict__ gate, float* __restrict__ out) {
  int i = blockIdx.x * 256 + threadIdx.x;
  if (i >= BL * DM) return;
  float g = gate[0];
  out[i] = g * ssm[i] + (1.0f - g) * att[i];
}

// ---------------------------------------------------------------------------
extern "C" void kernel_launch(void* const* d_in, const int* in_sizes, int n_in,
                              void* d_out, int out_size, void* d_ws, size_t ws_size,
                              hipStream_t stream) {
  (void)in_sizes; (void)n_in; (void)out_size; (void)ws_size;
  const float* x        = (const float*)d_in[0];
  const float* n1w      = (const float*)d_in[1];
  const float* n1b      = (const float*)d_in[2];
  const float* n2w      = (const float*)d_in[3];
  const float* n2b      = (const float*)d_in[4];
  const float* n3w      = (const float*)d_in[5];
  const float* n3b      = (const float*)d_in[6];
  const float* A_log    = (const float*)d_in[7];
  const float* Dp       = (const float*)d_in[8];
  const float* x_proj_w = (const float*)d_in[9];
  const float* dt_w     = (const float*)d_in[10];
  const float* dt_b     = (const float*)d_in[11];
  const float* conv_w   = (const float*)d_in[12];
  const float* conv_b   = (const float*)d_in[13];
  const float* wq       = (const float*)d_in[14];
  const float* wk       = (const float*)d_in[15];
  const float* wvp      = (const float*)d_in[16];
  const float* wo_w     = (const float*)d_in[17];
  const float* wo_b     = (const float*)d_in[18];
  const float* mlp_w1   = (const float*)d_in[19];
  const float* mlp_b1   = (const float*)d_in[20];
  const float* mlp_w2   = (const float*)d_in[21];
  const float* mlp_b2   = (const float*)d_in[22];
  const float* gate     = (const float*)d_in[23];
  float* out = (float*)d_out;

  char* base = (char*)d_ws;
  size_t off = 0;
  auto allocf = [&](size_t n) { float* p = (float*)(base + off);
                                off += ((n * 4 + 15) & ~(size_t)15); return p; };
  auto alloch = [&](size_t n) { _Float16* p = (_Float16*)(base + off);
                                off += ((n * 2 + 15) & ~(size_t)15); return p; };
  // f32 buffers
  float* xn       = allocf((size_t)BL * DM);
  float* xconv    = allocf((size_t)BL * DM);
  float* xdbl     = allocf((size_t)BL * XDBL_N);
  float* dtb      = allocf((size_t)BL * DM);
  float* ssm      = allocf((size_t)BL * DM);
  float* attn_out = allocf((size_t)BL * DM);
  float* combined = allocf((size_t)BL * DM);
  float* scores   = allocf((size_t)SEQ * SEQ);
  // f16 buffers
  _Float16* xn16     = alloch((size_t)BL * DM);
  _Float16* xconv16  = alloch((size_t)BL * DM);
  _Float16* xdbl16   = alloch((size_t)BL * XDBL_N);
  _Float16* q16      = alloch((size_t)BL * DM);
  _Float16* k16      = alloch((size_t)BL * DM);
  _Float16* vT16     = alloch((size_t)BL * DM);       // [b][h][j][m]
  _Float16* sm16     = alloch((size_t)SEQ * SEQ);
  _Float16* ao16     = alloch((size_t)BL * DM);
  _Float16* h116     = alloch((size_t)BL * MLPH);
  _Float16* xprojT   = alloch((size_t)XDBL_N * DM);   // 80 x 768
  _Float16* dtwT     = alloch((size_t)DM * 64);       // 768 x 64 (K padded 48->64)
  _Float16* wqT      = alloch((size_t)DM * DM);
  _Float16* wkT      = alloch((size_t)DM * DM);
  _Float16* wvT      = alloch((size_t)DM * DM);
  _Float16* woT      = alloch((size_t)DM * DM);
  _Float16* w1T      = alloch((size_t)MLPH * DM);
  _Float16* w2T      = alloch((size_t)DM * MLPH);
  (void)alloch(64 * 64);   // tail pad so edge-GEMM B over-reads stay in ws

  const dim3 blk(256);
  const dim3 gN768(DM / BN, BL / BM);   // (6,32)
  auto cgrid = [](long n) { return dim3((unsigned)((n + 255) / 256)); };

  // ---- one-time weight convert+transpose into N x K layouts ----
  cvtT_k<<<cgrid((long)XDBL_N * DM), blk, 0, stream>>>(x_proj_w, xprojT, DM, XDBL_N, DM);
  cvtT_k<<<cgrid((long)DM * 64), blk, 0, stream>>>(dt_w, dtwT, DTRANK, DM, 64);
  cvtT_k<<<cgrid((long)DM * DM), blk, 0, stream>>>(wq, wqT, DM, DM, DM);
  cvtT_k<<<cgrid((long)DM * DM), blk, 0, stream>>>(wk, wkT, DM, DM, DM);
  cvtT_k<<<cgrid((long)DM * DM), blk, 0, stream>>>(wvp, wvT, DM, DM, DM);
  cvtT_k<<<cgrid((long)DM * DM), blk, 0, stream>>>(wo_w, woT, DM, DM, DM);
  cvtT_k<<<cgrid((long)MLPH * DM), blk, 0, stream>>>(mlp_w1, w1T, DM, MLPH, DM);
  cvtT_k<<<cgrid((long)DM * MLPH), blk, 0, stream>>>(mlp_w2, w2T, MLPH, DM, MLPH);

  // ---------------- SSM branch ----------------
  ln_k<<<BL, blk, 0, stream>>>(x, n1w, n1b, xn, xn16);
  conv_k<<<cgrid((long)BL * DM), blk, 0, stream>>>(xn, conv_w, conv_b, xconv, xconv16);
  gemm_wmma_k<0, 2><<<dim3(1, 32), blk, 0, stream>>>(        // x_dbl (N=80)
      xconv16, xprojT, xdbl, xdbl16, nullptr, nullptr,
      XDBL_N, DM, DM, DM, XDBL_N, 1.0f);
  gemm_wmma_k<2, 0><<<gN768, blk, 0, stream>>>(              // dt = softplus, K padded 64
      xdbl16, dtwT, dtb, nullptr, dt_b, nullptr,
      DM, 64, XDBL_N, 64, DM, 1.0f);
  scan_k<<<dim3(DM / 256, BSZ), blk, 0, stream>>>(A_log, Dp, xdbl, dtb, xconv, x, ssm);

  // ---------------- attention branch ----------------
  ln_k<<<BL, blk, 0, stream>>>(x, n2w, n2b, xn, xn16);
  gemm_wmma_k<0, 1><<<gN768, blk, 0, stream>>>(xn16, wqT, nullptr, q16, nullptr, nullptr,
                                               DM, DM, DM, DM, DM, 1.0f);
  gemm_wmma_k<0, 1><<<gN768, blk, 0, stream>>>(xn16, wkT, nullptr, k16, nullptr, nullptr,
                                               DM, DM, DM, DM, DM, 1.0f);
  gemm_wmma_k<0, 3><<<gN768, blk, 0, stream>>>(xn16, wvT, nullptr, vT16, nullptr, nullptr,
                                               DM, DM, DM, DM, DM, 1.0f);

  const float sc = 0.10206207262f;   // 96^-0.5
  for (int b = 0; b < BSZ; ++b)
    for (int h = 0; h < NHEADS; ++h) {
      const _Float16* qp = q16 + (size_t)b * SEQ * DM + h * DK;
      const _Float16* kp = k16 + (size_t)b * SEQ * DM + h * DK;
      const _Float16* vp = vT16 + ((size_t)b * NHEADS + h) * DK * SEQ;
      _Float16*       op = ao16 + (size_t)b * SEQ * DM + h * DK;
      gemm_wmma_k<0, 0><<<dim3(SEQ / BN, SEQ / BM), blk, 0, stream>>>(
          qp, kp, scores, nullptr, nullptr, nullptr,
          SEQ, DK, DM, DM, SEQ, sc);
      softmax_k<<<SEQ, blk, 0, stream>>>(scores, sm16);
      gemm_wmma_k<0, 1><<<dim3(1, SEQ / BM), blk, 0, stream>>>(  // N=96
          sm16, vp, nullptr, op, nullptr, nullptr,
          DK, SEQ, SEQ, SEQ, DM, 1.0f);
    }
  gemm_wmma_k<4, 0><<<gN768, blk, 0, stream>>>(              // x + ao@Wo + b
      ao16, woT, attn_out, nullptr, wo_b, x, DM, DM, DM, DM, DM, 1.0f);

  // ---------------- combine + MLP ----------------
  combine_k<<<cgrid((long)BL * DM), blk, 0, stream>>>(ssm, attn_out, gate, combined);
  ln_k<<<BL, blk, 0, stream>>>(combined, n3w, n3b, xn, xn16);
  gemm_wmma_k<3, 1><<<dim3(MLPH / BN, BL / BM), blk, 0, stream>>>(  // gelu
      xn16, w1T, nullptr, h116, mlp_b1, nullptr,
      MLPH, DM, DM, DM, MLPH, 1.0f);
  gemm_wmma_k<4, 0><<<gN768, blk, 0, stream>>>(              // + combined residual
      h116, w2T, out, nullptr, mlp_b2, combined,
      DM, MLPH, MLPH, MLPH, DM, 1.0f);
}
